// LocalSphereAttention_34428457845052
// MI455X (gfx1250) — matrix operand order repaired and checked by reading, not compiled, not run on hardware
//
#include <hip/hip_runtime.h>
#include <hip/hip_bf16.h>

#define DIM    128
#define NPTS   8192
#define NBATCH 2
#define NHEADS 4
#define HEADD  32
#define KNN    16
#define BIASH  32

typedef __attribute__((ext_vector_type(16))) _Float16 v16h;
typedef __attribute__((ext_vector_type(8)))  _Float16 v8h;
typedef __attribute__((ext_vector_type(8)))  float    v8f;
typedef __attribute__((ext_vector_type(4)))  unsigned int v4u;
typedef __attribute__((ext_vector_type(8)))  int v8i;
typedef __attribute__((ext_vector_type(4)))  int v4i;

// ---------------------------------------------------------------------------
// Kernel 1: KNN via WMMA distance tiles.
// Each wave owns 16 query points (columns of the WMMA C tile) and streams all
// candidate tiles of its batch. score(M=cand,N=query) = |c|^2 - 2 c.q, which
// has the same per-query ordering as the true squared distance.
// Lanes l and l+16 track complementary candidate rows for the same query and
// are merged through LDS at the end.
// ---------------------------------------------------------------------------
__global__ __launch_bounds__(256) void lsa_knn_kernel(const float* __restrict__ xyz,
                                                      int* __restrict__ knn_idx)
{
  __shared__ float md[256][KNN];
  __shared__ int   mi[256][KNN];

  const int lane = threadIdx.x & 31;
  const int wave = threadIdx.x >> 5;
  const int col  = lane & 15;
  const int half = lane >> 4;

  const int gwave = blockIdx.x * 8 + wave;          // 0 .. B*N/16-1
  const int b     = gwave / (NPTS / 16);
  const int qt    = gwave % (NPTS / 16);
  const int qbase = qt * 16;

  const float sel = (half == 0) ? 1.0f : 0.0f;      // only K=0..3 carry data

  // B fragment (queries, K x N = 32 x 16). Lanes 0-15 hold K0..7 & K16..23 of column.
  const int qp = b * NPTS + qbase + col;
  const float qx = xyz[qp * 3 + 0];
  const float qy = xyz[qp * 3 + 1];
  const float qz = xyz[qp * 3 + 2];

  v16h bf;
  #pragma unroll
  for (int i = 0; i < 16; ++i) bf[i] = (_Float16)0.0f;
  bf[0] = (_Float16)(-2.0f * qx * sel);
  bf[1] = (_Float16)(-2.0f * qy * sel);
  bf[2] = (_Float16)(-2.0f * qz * sel);
  bf[3] = (_Float16)sel;

  float bd[KNN]; int bi[KNN];
  #pragma unroll
  for (int j = 0; j < KNN; ++j) { bd[j] = 3.0e38f; bi[j] = 0; }

  for (int ct = 0; ct < NPTS / 16; ++ct) {
    const int cp = b * NPTS + ct * 16 + col;
    const float cx = xyz[cp * 3 + 0];
    const float cy = xyz[cp * 3 + 1];
    const float cz = xyz[cp * 3 + 2];
    const float c2 = cx * cx + cy * cy + cz * cz;

    // speculative prefetch of next candidate tile (L2-resident anyway)
    const int pf = b * NPTS + ((ct + 1) & (NPTS / 16 - 1)) * 16 + col;
    __builtin_prefetch(&xyz[pf * 3], 0, 0);

    // A fragment (candidates, M x K = 16 x 32). Lanes 0-15 hold K0..7 in
    // halfs 0..3 -> only those carry the payload, rest are zero.
    v16h af;
    #pragma unroll
    for (int i = 0; i < 16; ++i) af[i] = (_Float16)0.0f;
    af[0] = (_Float16)(cx * sel);
    af[1] = (_Float16)(cy * sel);
    af[2] = (_Float16)(cz * sel);
    af[3] = (_Float16)(c2 * sel);

    v8f cz8;
    #pragma unroll
    for (int i = 0; i < 8; ++i) cz8[i] = 0.0f;

    v8f acc = __builtin_amdgcn_wmma_f32_16x16x32_f16(false, af, false, bf,
                                                     (short)0, cz8, false, false);

    // C layout: lane < 16 -> rows 0..7, lane >= 16 -> rows 8..15; col = lane&15
    #pragma unroll
    for (int p = 0; p < 8; ++p) {
      const float d  = acc[p];
      const int   ci = ct * 16 + half * 8 + p;
      if (d < bd[KNN - 1]) {
        int pos = 0;
        #pragma unroll
        for (int j = 0; j < KNN; ++j) pos += (bd[j] <= d) ? 1 : 0;
        #pragma unroll
        for (int j = KNN - 1; j > 0; --j) {
          const bool sh = (j > pos);
          bd[j] = sh ? bd[j - 1] : bd[j];
          bi[j] = sh ? bi[j - 1] : bi[j];
        }
        #pragma unroll
        for (int j = 0; j < KNN; ++j)
          if (j == pos) { bd[j] = d; bi[j] = ci; }
      }
    }
  }

  #pragma unroll
  for (int j = 0; j < KNN; ++j) { md[threadIdx.x][j] = bd[j]; mi[threadIdx.x][j] = bi[j]; }
  __syncthreads();

  if (lane < 16) {
    const int l0 = wave * 32 + lane;
    const int l1 = l0 + 16;
    int ia = 0, ib2 = 0;
    int* op = &knn_idx[(size_t)(b * NPTS + qbase + lane) * KNN];
    #pragma unroll
    for (int t = 0; t < KNN; ++t) {
      const float da = (ia  < KNN) ? md[l0][ia]  : 3.0e38f;
      const float db = (ib2 < KNN) ? md[l1][ib2] : 3.0e38f;
      const bool  ta = (da <= db);
      op[t] = ta ? mi[l0][ia] : mi[l1][ib2];
      ia  += ta ? 1 : 0;
      ib2 += ta ? 0 : 1;
    }
  }
}

// ---------------------------------------------------------------------------
// Kernel 2: out[M,128] = A[M,128] @ W[128,128] + bias, f16 WMMA, f32 accum.
// The 128x128 f32 A-tile is fetched by the Tensor Data Mover directly into
// LDS with hardware row padding (pad_interval=128 DW, pad_amount=4 DW ->
// 132-float rows for bank spreading). Fragments convert f32->f16 in regs.
// W is staged transposed as f16 by the threads (TDM cannot convert).
// ---------------------------------------------------------------------------
__global__ __launch_bounds__(256) void lsa_gemm128_kernel(const float* __restrict__ A,
                                                          const float* __restrict__ W,
                                                          const float* __restrict__ bias,
                                                          float* __restrict__ out)
{
  __shared__ __align__(16) float    sXf[128 * 132];   // TDM destination (padded)
  __shared__ __align__(16) _Float16 sWT[128 * 136];

  const int    tid     = threadIdx.x;
  const size_t rowBase = (size_t)blockIdx.x * 128;

  // --- TDM: async DMA of the A tile (f32) into LDS, one issue per block ---
  if (tid < 32) {
    const unsigned long long gaddr =
        (unsigned long long)(uintptr_t)(A + rowBase * DIM);
    const unsigned ldsAddr = (unsigned)(uintptr_t)&sXf[0];

    v4u g0;
    g0[0] = 1u;                                  // count=1, user descriptor
    g0[1] = ldsAddr;                             // lds_addr
    g0[2] = (unsigned)(gaddr & 0xffffffffull);   // global_addr[31:0]
    g0[3] = (unsigned)((gaddr >> 32) & 0x01ffffffull) | (2u << 30); // type=2

    v8i g1;
    g1[0] = (2 << 16) | (1 << 20) | (6 << 22) | (3 << 25);
           // data_size=4B, pad_enable, pad_interval=128DW, pad_amount=4DW
    g1[1] = (DIM & 0xffff) << 16;                // tensor_dim0 = 128 (lo16)
    g1[2] = ((NBATCH * NPTS) & 0xffff) << 16;    // tensor_dim1 = 16384 (lo16)
    g1[3] = (DIM << 16);                         // tile_dim0 = 128
    g1[4] = 128;                                 // tile_dim1 = 128, tile_dim2 = 0
    g1[5] = DIM;                                 // tensor_dim0_stride = 128
    g1[6] = 0;                                   // dim1_stride lo16 = 0
    g1[7] = (DIM * NBATCH * NPTS) >> 16;         // dim1_stride hi32 = 32

    v4i g2;
    g2[0] = 1;                                   // tensor_dim2 = 1 (unused)
    g2[1] = 0; g2[2] = 0; g2[3] = 0;             // tile_dim3 = 0

    v4i g3;
    g3[0] = 0; g3[1] = 0; g3[2] = 0; g3[3] = 0;

#if __has_include(<hip/amd_detail/amd_gfx1250_TDM.h>)
    v8i gpad;
    #pragma unroll
    for (int i = 0; i < 8; ++i) gpad[i] = 0;
    __builtin_amdgcn_tensor_load_to_lds(g0, g1, g2, g3, gpad, 0);
#else
    __builtin_amdgcn_tensor_load_to_lds(g0, g1, g2, g3, 0);
#endif
    __builtin_amdgcn_s_wait_tensorcnt(0);
  }

  // --- stage W transposed (f32 -> f16, column-major in LDS) ---
  #pragma unroll
  for (int i = 0; i < 16; ++i) {
    const int chunk = tid + 256 * i;
    const int k  = chunk >> 5;
    const int c4 = (chunk & 31) * 4;
    const float4 v = *(const float4*)&W[k * DIM + c4];
    sWT[(c4 + 0) * 136 + k] = (_Float16)v.x;
    sWT[(c4 + 1) * 136 + k] = (_Float16)v.y;
    sWT[(c4 + 2) * 136 + k] = (_Float16)v.z;
    sWT[(c4 + 3) * 136 + k] = (_Float16)v.w;
  }
  __syncthreads();

  const int wave = tid >> 5;
  const int lane = tid & 31;
  const int half = lane >> 4;
  const int l16  = lane & 15;
  const int mrow = wave * 16 + l16;

  v8f acc[8];
  #pragma unroll
  for (int nt = 0; nt < 8; ++nt) {
    #pragma unroll
    for (int i = 0; i < 8; ++i) acc[nt][i] = 0.0f;
  }

  #pragma unroll
  for (int kt = 0; kt < 4; ++kt) {
    const int kb = kt * 32;
    // A frag: lanes 0-15 K0..7 & K16..23 ; lanes 16-31 K8..15 & K24..31
    const float4 a0 = *(const float4*)&sXf[mrow * 132 + kb + half * 8];
    const float4 a1 = *(const float4*)&sXf[mrow * 132 + kb + half * 8 + 4];
    const float4 a2 = *(const float4*)&sXf[mrow * 132 + kb + 16 + half * 8];
    const float4 a3 = *(const float4*)&sXf[mrow * 132 + kb + 16 + half * 8 + 4];
    v16h afrag;
    afrag[0]  = (_Float16)a0.x; afrag[1]  = (_Float16)a0.y;
    afrag[2]  = (_Float16)a0.z; afrag[3]  = (_Float16)a0.w;
    afrag[4]  = (_Float16)a1.x; afrag[5]  = (_Float16)a1.y;
    afrag[6]  = (_Float16)a1.z; afrag[7]  = (_Float16)a1.w;
    afrag[8]  = (_Float16)a2.x; afrag[9]  = (_Float16)a2.y;
    afrag[10] = (_Float16)a2.z; afrag[11] = (_Float16)a2.w;
    afrag[12] = (_Float16)a3.x; afrag[13] = (_Float16)a3.y;
    afrag[14] = (_Float16)a3.z; afrag[15] = (_Float16)a3.w;

    #pragma unroll
    for (int nt = 0; nt < 8; ++nt) {
      const int c = nt * 16 + l16;
      // B frag: lanes 0-15 K0..7 & K16..23 of column c; lanes 16-31 K8..15 & K24..31
      const v8h b_lo = *(const v8h*)&sWT[c * 136 + kb + half * 8];
      const v8h b_hi = *(const v8h*)&sWT[c * 136 + kb + 16 + half * 8];
      v16h bfrag;
      #pragma unroll
      for (int i = 0; i < 8; ++i) { bfrag[i] = b_lo[i]; bfrag[i + 8] = b_hi[i]; }
      acc[nt] = __builtin_amdgcn_wmma_f32_16x16x32_f16(false, afrag, false, bfrag,
                                                       (short)0, acc[nt], false, false);
    }
  }

  #pragma unroll
  for (int nt = 0; nt < 8; ++nt) {
    const int   c  = nt * 16 + l16;
    const float bv = bias[c];
    #pragma unroll
    for (int p = 0; p < 8; ++p) {
      const size_t m = rowBase + wave * 16 + half * 8 + p;
      out[m * DIM + c] = acc[nt][p] + bv;
    }
  }
}

// ---------------------------------------------------------------------------
// Kernel 3: gathered 16-neighbor attention + 3->32->4 bias MLP.
// One thread per (point, head); k/v are L2-resident (16 MB << 192 MB).
// ---------------------------------------------------------------------------
__global__ __launch_bounds__(256) void lsa_attn_kernel(const float* __restrict__ qf,
                                                       const float* __restrict__ kf,
                                                       const float* __restrict__ vf,
                                                       const float* __restrict__ xyz,
                                                       const int* __restrict__ knn_idx,
                                                       const float* __restrict__ W1,
                                                       const float* __restrict__ b1,
                                                       const float* __restrict__ W2,
                                                       const float* __restrict__ b2,
                                                       float* __restrict__ att)
{
  __shared__ float sW1[3 * BIASH];
  __shared__ float sb1[BIASH];
  __shared__ float sW2[BIASH * NHEADS];
  __shared__ float sb2[NHEADS];

  const int tid = threadIdx.x;
  if (tid < 3 * BIASH)      sW1[tid] = W1[tid];
  if (tid < BIASH)          sb1[tid] = b1[tid];
  if (tid < BIASH * NHEADS) sW2[tid] = W2[tid];
  if (tid < NHEADS)         sb2[tid] = b2[tid];
  __syncthreads();

  const int t  = blockIdx.x * 256 + tid;   // ((b*N)+n)*H + h
  const int h  = t & (NHEADS - 1);
  const int bn = t >> 2;                   // b*N + n
  const int b  = bn >> 13;                 // / 8192

  float qv[HEADD];
  {
    const float4* p4 = (const float4*)&qf[(size_t)bn * DIM + h * HEADD];
    #pragma unroll
    for (int i = 0; i < 8; ++i) {
      const float4 v = p4[i];
      qv[4*i+0] = v.x; qv[4*i+1] = v.y; qv[4*i+2] = v.z; qv[4*i+3] = v.w;
    }
  }
  const float px = xyz[(size_t)bn * 3 + 0];
  const float py = xyz[(size_t)bn * 3 + 1];
  const float pz = xyz[(size_t)bn * 3 + 2];

  float sc[KNN]; int ni[KNN];
  #pragma unroll
  for (int j = 0; j < KNN; ++j) {
    const int n2 = knn_idx[(size_t)bn * KNN + j];
    ni[j] = n2;
    const size_t nb = (size_t)b * NPTS + n2;
    const float4* kp = (const float4*)&kf[nb * DIM + h * HEADD];
    float s = 0.0f;
    #pragma unroll
    for (int i = 0; i < 8; ++i) {
      const float4 v = kp[i];
      s += qv[4*i]*v.x + qv[4*i+1]*v.y + qv[4*i+2]*v.z + qv[4*i+3]*v.w;
    }
    const float rx = px - xyz[nb * 3 + 0];
    const float ry = py - xyz[nb * 3 + 1];
    const float rz = pz - xyz[nb * 3 + 2];
    float bh = sb2[h];
    #pragma unroll
    for (int m = 0; m < BIASH; ++m) {
      float hid = rx * sW1[m] + ry * sW1[BIASH + m] + rz * sW1[2 * BIASH + m] + sb1[m];
      hid = fmaxf(hid, 0.0f);
      bh += hid * sW2[m * NHEADS + h];
    }
    sc[j] = s * 0.17677669529663689f + bh;   // 1/sqrt(32)
  }

  float mx = -3.0e38f;
  #pragma unroll
  for (int j = 0; j < KNN; ++j) mx = fmaxf(mx, sc[j]);
  float se = 0.0f;
  #pragma unroll
  for (int j = 0; j < KNN; ++j) { sc[j] = __expf(sc[j] - mx); se += sc[j]; }
  const float inv = 1.0f / se;

  float o[HEADD];
  #pragma unroll
  for (int d = 0; d < HEADD; ++d) o[d] = 0.0f;
  #pragma unroll
  for (int j = 0; j < KNN; ++j) {
    const float  w  = sc[j] * inv;
    const size_t nb = (size_t)b * NPTS + ni[j];
    const float4* vp = (const float4*)&vf[nb * DIM + h * HEADD];
    #pragma unroll
    for (int i = 0; i < 8; ++i) {
      const float4 v = vp[i];
      o[4*i] += w*v.x; o[4*i+1] += w*v.y; o[4*i+2] += w*v.z; o[4*i+3] += w*v.w;
    }
  }
  float* op = &att[(size_t)bn * DIM + h * HEADD];
  #pragma unroll
  for (int d = 0; d < HEADD; ++d) op[d] = o[d];
}

// ---------------------------------------------------------------------------
extern "C" void kernel_launch(void* const* d_in, const int* in_sizes, int n_in,
                              void* d_out, int out_size, void* d_ws, size_t ws_size,
                              hipStream_t stream) {
  (void)in_sizes; (void)n_in; (void)out_size; (void)ws_size;

  const float* x   = (const float*)d_in[0];
  const float* xyz = (const float*)d_in[1];
  const float* Wq  = (const float*)d_in[2];
  const float* bq  = (const float*)d_in[3];
  const float* Wk  = (const float*)d_in[4];
  const float* bk  = (const float*)d_in[5];
  const float* Wv  = (const float*)d_in[6];
  const float* bv  = (const float*)d_in[7];
  const float* Wo  = (const float*)d_in[8];
  const float* bo  = (const float*)d_in[9];
  const float* W1  = (const float*)d_in[10];
  const float* b1  = (const float*)d_in[11];
  const float* W2  = (const float*)d_in[12];
  const float* b2  = (const float*)d_in[13];
  float* out = (float*)d_out;

  const size_t BN = (size_t)NBATCH * NPTS;            // 16384 points
  char* ws = (char*)d_ws;
  int*   idxb = (int*)ws;
  size_t off  = (BN * KNN * sizeof(int) + 255) & ~(size_t)255;
  float* qb = (float*)(ws + off);
  float* kb = qb + BN * DIM;
  float* vb = kb + BN * DIM;
  float* ab = vb + BN * DIM;

  const dim3 blk(256);
  const dim3 gTiles((unsigned)(BN / 128));            // 128 blocks

  lsa_knn_kernel<<<gTiles, blk, 0, stream>>>(xyz, idxb);
  lsa_gemm128_kernel<<<gTiles, blk, 0, stream>>>(x, Wq, bq, qb);
  lsa_gemm128_kernel<<<gTiles, blk, 0, stream>>>(x, Wk, bk, kb);
  lsa_gemm128_kernel<<<gTiles, blk, 0, stream>>>(x, Wv, bv, vb);
  lsa_attn_kernel<<<dim3((unsigned)(BN * NHEADS / 256)), blk, 0, stream>>>(
      qb, kb, vb, xyz, idxb, W1, b1, W2, b2, ab);
  lsa_gemm128_kernel<<<gTiles, blk, 0, stream>>>(ab, Wo, bo, out);
}
